// NetVLAD_13280038880028
// MI455X (gfx1250) — compile-verified, matrix operand order checked by reading, not run on hardware
//
#include <hip/hip_runtime.h>

// ---------------------------------------------------------------------------
// NetVLAD for MI455X (gfx1250): bf16 WMMA for the three GEMMs
//   logits = conv_w[64x128] @ x[128x4096]  (per image)
//   vlad   = a[64x4096]    @ xT[4096x128]  (per image)
//   out    = V2[32x8192]   @ mlp_wT[8192x256]
// All f32->bf16 conversion hoisted out of K-loops; k2 streams pre-converted
// bf16 x through an async-to-LDS ping-pong buffer.
// ---------------------------------------------------------------------------

typedef __attribute__((ext_vector_type(16))) __bf16 v16bf;
typedef __attribute__((ext_vector_type(8)))  float  v8f;
typedef __attribute__((ext_vector_type(4)))  int    v4i;

union BF16Frag {
    unsigned int u[8];
    v16bf        v;
};

#define EPS 1e-12f

// ---- feature probes -------------------------------------------------------
#ifdef __has_builtin
#if __has_builtin(__builtin_amdgcn_global_load_async_to_lds_b128) && \
    __has_builtin(__builtin_amdgcn_s_wait_asynccnt)
#define USE_ASYNC_LDS 1
#endif
#endif

// f32 -> bf16 (RNE) helpers
__device__ __forceinline__ unsigned short f2bf(float f) {
    unsigned int a = __builtin_bit_cast(unsigned int, f);
    return (unsigned short)((a + 0x7FFFu + ((a >> 16) & 1u)) >> 16);
}
__device__ __forceinline__ unsigned int pack_bf2(float lo, float hi) {
    unsigned int a = __builtin_bit_cast(unsigned int, lo);
    unsigned int b = __builtin_bit_cast(unsigned int, hi);
    a = (a + 0x7FFFu + ((a >> 16) & 1u)) >> 16;
    b = (b + 0x7FFFu + ((b >> 16) & 1u)) >> 16;
    return a | (b << 16);
}

// 16B global -> LDS staging copy (async-to-LDS path when available)
__device__ __forceinline__ void stage16(const void* g, void* l) {
#if defined(USE_ASYNC_LDS)
    __builtin_amdgcn_global_load_async_to_lds_b128(
        (__attribute__((address_space(1))) v4i*)(uintptr_t)g,
        (__attribute__((address_space(3))) v4i*)(unsigned int)(uintptr_t)l,
        0, 0);
#else
    *reinterpret_cast<v4i*>(l) = *reinterpret_cast<const v4i*>(g);
#endif
}
__device__ __forceinline__ void stage_wait0() {
#if defined(USE_ASYNC_LDS)
    __builtin_amdgcn_s_wait_asynccnt(0);
#endif
}
__device__ __forceinline__ void stage_wait2() {   // allow newest batch in flight
#if defined(USE_ASYNC_LDS)
    __builtin_amdgcn_s_wait_asynccnt(2);
#endif
}

// 16-bit A/B fragment K-offset for VGPR j (pairs): lane-group kg = (lane/16)*8
__device__ __forceinline__ int koff(int lane, int j) {
    int kg = (lane >> 4) * 8;
    return (j < 4) ? (kg + 2 * j) : (16 + kg + 2 * (j - 4));
}

// ---------------------------------------------------------------------------
// Prep kernel: mlp_w -> bf16 (blocks 0..511), conv_w -> bf16 (block 512),
// zero asum (block 513). One-shot, hoists all weight conversion.
// ---------------------------------------------------------------------------
__global__ void kp_prep(const float* __restrict__ conv_w,
                        const float* __restrict__ mlp_w,
                        unsigned int* __restrict__ cwb,
                        unsigned int* __restrict__ mwb,
                        float* __restrict__ asum) {
    const int b   = blockIdx.x;
    const int tid = threadIdx.x;
    if (b < 512) {
#pragma unroll
        for (int i = 0; i < 8; ++i) {
            const size_t d = (size_t)b * 2048 + i * 256 + tid;   // dword idx
            mwb[d] = pack_bf2(mlp_w[2 * d], mlp_w[2 * d + 1]);
        }
    } else if (b == 512) {
#pragma unroll
        for (int i = 0; i < 16; ++i) {
            const int d = i * 256 + tid;                          // 4096 dwords
            cwb[d] = pack_bf2(conv_w[2 * d], conv_w[2 * d + 1]);
        }
    } else {
#pragma unroll
        for (int i = 0; i < 8; ++i) asum[i * 256 + tid] = 0.0f;
    }
}

// ---------------------------------------------------------------------------
// Kernel 1: logits (WMMA bf16) + bias + softmax over K=64, per 32-pixel slab.
// grid = (128 slabs, 32 images), block = 256 (8 waves).
// Also emits the bf16 copy of x ([n][c][p]) consumed by k2.
// ---------------------------------------------------------------------------
__global__ void k1_logits_softmax(const float* __restrict__ x,
                                  const unsigned int* __restrict__ cwb,
                                  const float* __restrict__ conv_b,
                                  unsigned short* __restrict__ a_ws,
                                  float* __restrict__ asum,
                                  unsigned int* __restrict__ xbf) {
    const int n     = blockIdx.y;
    const int pbase = blockIdx.x * 32;
    const int tid   = threadIdx.x;
    const int wave  = tid >> 5;
    const int lane  = tid & 31;
    const int mtile = wave & 3;
    const int ntile = wave >> 2;

    __shared__ float        xs[128 * 32];     // f32 staging, [c][p]
    __shared__ unsigned int xb[32 * 65];      // bf16 pairs, [p][c/2], padded
    __shared__ float        lgt[64 * 33];     // logits, padded rows
    __shared__ float        psum[32];

    // ---- stage x slab: 4096 floats, 4x b128 per thread ----
    const float* xbase = x + (size_t)n * 128 * 4096 + pbase;
#pragma unroll
    for (int i = 0; i < 4; ++i) {
        const int f = i * 256 + tid;          // 1024 groups of 4 floats
        const int c = f >> 3;
        const int g = (f & 7) * 4;
        stage16(xbase + (size_t)c * 4096 + g, &xs[c * 32 + g]);
    }
    stage_wait0();
    __syncthreads();

    // ---- convert to packed bf16 [p][c-pair] for the B fragments ----
#pragma unroll
    for (int i = 0; i < 8; ++i) {
        const int d = i * 256 + tid;          // 2048 dwords
        const int w = d >> 5;                 // c-pair index 0..63
        const int p = d & 31;
        xb[p * 65 + w] = pack_bf2(xs[(2 * w) * 32 + p], xs[(2 * w + 1) * 32 + p]);
    }
    // ---- also emit bf16 x in [c][p] layout (p-pairs) for k2 ----
#pragma unroll
    for (int i = 0; i < 8; ++i) {
        const int d  = i * 256 + tid;         // 2048 dwords
        const int c  = d >> 4;
        const int pq = d & 15;
        xbf[(size_t)(n * 128 + c) * 2048 + (pbase >> 1) + pq] =
            pack_bf2(xs[c * 32 + 2 * pq], xs[c * 32 + 2 * pq + 1]);
    }
    __syncthreads();

    // ---- WMMA: conv_w (bf16 A from ws) x x-slab (B from LDS) ----
    v8f acc = {};
    const int krow = mtile * 16 + (lane & 15);
    const int pl   = ntile * 16 + (lane & 15);
    for (int c0 = 0; c0 < 128; c0 += 32) {
        BF16Frag fa, fb;
#pragma unroll
        for (int j = 0; j < 8; ++j) {
            const int c = c0 + koff(lane, j);
            fa.u[j] = cwb[krow * 64 + (c >> 1)];
            fb.u[j] = xb[pl * 65 + (c >> 1)];
        }
        acc = __builtin_amdgcn_wmma_f32_16x16x32_bf16(false, fa.v, false, fb.v,
                                                      (short)0, acc, false, false);
    }

    // scatter logits (+bias) into LDS as [k][p_local]
#pragma unroll
    for (int r = 0; r < 8; ++r) {
        const int k = mtile * 16 + r + 8 * (lane >> 4);
        lgt[k * 33 + pl] = acc[r] + conv_b[k];
    }
    __syncthreads();

    // softmax over the 64 clusters, one pixel per thread
    if (tid < 32) {
        const int p = tid;
        float m = -3.4e38f;
        for (int k = 0; k < 64; ++k) m = fmaxf(m, lgt[k * 33 + p]);
        float s = 0.0f;
        for (int k = 0; k < 64; ++k) {
            float e = __expf(lgt[k * 33 + p] - m);
            lgt[k * 33 + p] = e;
            s += e;
        }
        psum[p] = s;
    }
    __syncthreads();

    // per-k: normalize, emit bf16 row (contiguous 64B store), partial asum
    if (tid < 64) {
        const int k = tid;
        float partial = 0.0f;
        unsigned short* arow = a_ws + ((size_t)n * 64 + k) * 4096 + pbase;
#pragma unroll
        for (int p = 0; p < 32; ++p) {
            float av = lgt[k * 33 + p] / psum[p];
            partial += av;
            arow[p] = f2bf(av);
        }
        atomicAdd(&asum[n * 64 + k], partial);
    }
}

// ---------------------------------------------------------------------------
// Kernel 2: vlad[n][k][c] = sum_p a[k,p] * x[c,p]   (WMMA, inner dim P=4096)
// grid = 32 images, block = 256 (8 waves).
// bf16 x slab async-staged into a ping-pong LDS buffer; zero conversion work.
// Row stride 20 dwords (80B): 16B-aligned async dests, conflict-free reads.
// ---------------------------------------------------------------------------
__global__ void k2_vlad(const unsigned short* __restrict__ a_ws,
                        const unsigned short* __restrict__ xbf,
                        float* __restrict__ vlad) {
    const int n     = blockIdx.x;
    const int tid   = threadIdx.x;
    const int wave  = tid >> 5;
    const int lane  = tid & 31;
    const int mtile = wave & 3;
    const int cbase = (wave >> 2) * 64;

    __shared__ unsigned int xb2[2][128 * 20];

    v8f acc[4] = {{}, {}, {}, {}};
    const int krow = mtile * 16 + (lane & 15);
    const unsigned short* arow = a_ws + ((size_t)n * 64 + krow) * 4096;
    const unsigned short* xrow = xbf + (size_t)n * 128 * 4096;

    auto stage_slab = [&](int buf, int p0) {
#pragma unroll
        for (int i = 0; i < 2; ++i) {
            const int f  = i * 256 + tid;     // 512 chunks of 16B
            const int c  = f >> 2;
            const int ch = f & 3;
            stage16(xrow + (size_t)c * 4096 + p0 + ch * 8,
                    &xb2[buf][c * 20 + ch * 4]);
        }
    };

    stage_slab(0, 0);
    for (int i = 0; i < 128; ++i) {
        const int p0 = i * 32;
        // issue next slab (wraps harmlessly on last iteration)
        stage_slab((i + 1) & 1, ((i + 1) & 127) * 32);
        stage_wait2();                        // current slab complete
        __syncthreads();

        const unsigned int* xcur = xb2[i & 1];

        // A fragment: bf16 pairs p-consecutive in a_ws -> direct dword loads
        BF16Frag fa;
#pragma unroll
        for (int j = 0; j < 8; ++j)
            fa.u[j] = *reinterpret_cast<const unsigned int*>(arow + p0 + koff(lane, j));
        __builtin_prefetch(arow + p0 + 1024, 0, 1);   // stream-ahead hint

#pragma unroll
        for (int t = 0; t < 4; ++t) {
            const int c = cbase + t * 16 + (lane & 15);
            BF16Frag fb;
#pragma unroll
            for (int j = 0; j < 8; ++j)
                fb.u[j] = xcur[c * 20 + (koff(lane, j) >> 1)];
            acc[t] = __builtin_amdgcn_wmma_f32_16x16x32_bf16(false, fa.v, false, fb.v,
                                                             (short)0, acc[t], false, false);
        }
        __syncthreads();                      // consumers done before overwrite
    }

#pragma unroll
    for (int t = 0; t < 4; ++t) {
        const int c = cbase + t * 16 + (lane & 15);
#pragma unroll
        for (int r = 0; r < 8; ++r) {
            const int k = mtile * 16 + r + 8 * (lane >> 4);
            vlad[((size_t)n * 64 + k) * 128 + c] = acc[t][r];
        }
    }
}

// ---------------------------------------------------------------------------
// Kernel 3: subtract asum[k]*centers, intra-norm over C, global L2 norm,
// write bf16 V2[n][8192]. grid = 32, block = 256.
// ---------------------------------------------------------------------------
__global__ void k3_finalize(const float* __restrict__ vlad,
                            const float* __restrict__ asum,
                            const float* __restrict__ centers,
                            unsigned short* __restrict__ v2) {
    const int n   = blockIdx.x;
    const int tid = threadIdx.x;
    __shared__ float vbuf[8192];
    __shared__ float rsumsq[64];
    __shared__ float rnorm[64];
    __shared__ float gnorm_s;

#pragma unroll
    for (int i = 0; i < 32; ++i) {
        const int idx = i * 256 + tid;
        const int k = idx >> 7, c = idx & 127;
        vbuf[idx] = vlad[(size_t)n * 8192 + idx] - asum[n * 64 + k] * centers[k * 128 + c];
    }
    __syncthreads();

    if (tid < 64) {
        float s = 0.0f;
        for (int c = 0; c < 128; ++c) {
            float v = vbuf[tid * 128 + c];
            s += v * v;
        }
        rsumsq[tid] = s;
        rnorm[tid]  = fmaxf(sqrtf(s), EPS);
    }
    __syncthreads();

    if (tid == 0) {
        float g = 0.0f;
        for (int k = 0; k < 64; ++k) {
            float rn = rnorm[k];
            g += rsumsq[k] / (rn * rn);
        }
        gnorm_s = fmaxf(sqrtf(g), EPS);
    }
    __syncthreads();

    const float gn = gnorm_s;
#pragma unroll
    for (int i = 0; i < 32; ++i) {
        const int idx = i * 256 + tid;
        const int k = idx >> 7;
        v2[(size_t)n * 8192 + idx] = f2bf(vbuf[idx] / rnorm[k] / gn);
    }
}

// ---------------------------------------------------------------------------
// Kernel 4: out_pre[32x256] = V2[32x8192] @ mlp_w^T + b   (WMMA bf16)
// grid = 32 single-wave blocks; both A and B are pre-packed bf16 dwords.
// ---------------------------------------------------------------------------
__global__ void k4_mlp(const unsigned short* __restrict__ v2,
                       const unsigned int* __restrict__ mwb,
                       const float* __restrict__ mlp_b,
                       float* __restrict__ out_pre) {
    const int b     = blockIdx.x;
    const int lane  = threadIdx.x & 31;
    const int mtile = b & 1;
    const int ntile = b >> 1;

    v8f acc = {};
    const int img  = mtile * 16 + (lane & 15);
    const int ocol = ntile * 16 + (lane & 15);
    const unsigned short* vrow = v2 + (size_t)img * 8192;
    const unsigned int*   wrow = mwb + (size_t)ocol * 4096;

    for (int k0 = 0; k0 < 8192; k0 += 32) {
        BF16Frag fa, fb;
#pragma unroll
        for (int j = 0; j < 8; ++j) {
            const int kk = k0 + koff(lane, j);
            fa.u[j] = *reinterpret_cast<const unsigned int*>(vrow + kk);
            fb.u[j] = wrow[kk >> 1];
        }
        acc = __builtin_amdgcn_wmma_f32_16x16x32_bf16(false, fa.v, false, fb.v,
                                                      (short)0, acc, false, false);
    }

#pragma unroll
    for (int r = 0; r < 8; ++r) {
        const int im = mtile * 16 + r + 8 * (lane >> 4);
        const int o  = ntile * 16 + (lane & 15);
        out_pre[(size_t)im * 256 + o] = acc[r] + mlp_b[o];
    }
}

// ---------------------------------------------------------------------------
// Kernel 5: final per-image L2 normalize -> d_out. grid = 32, block = 256.
// ---------------------------------------------------------------------------
__global__ void k5_norm(const float* __restrict__ out_pre, float* __restrict__ out) {
    const int n   = blockIdx.x;
    const int tid = threadIdx.x;
    __shared__ float red[256];
    const float v = out_pre[n * 256 + tid];
    red[tid] = v * v;
    __syncthreads();
    for (int s = 128; s > 0; s >>= 1) {
        if (tid < s) red[tid] += red[tid + s];
        __syncthreads();
    }
    out[n * 256 + tid] = v / fmaxf(sqrtf(red[0]), EPS);
}

// ---------------------------------------------------------------------------
extern "C" void kernel_launch(void* const* d_in, const int* in_sizes, int n_in,
                              void* d_out, int out_size, void* d_ws, size_t ws_size,
                              hipStream_t stream) {
    const float* x       = (const float*)d_in[0];  // [32,128,64,64]
    const float* centers = (const float*)d_in[1];  // [64,128]
    const float* conv_w  = (const float*)d_in[2];  // [64,128]
    const float* conv_b  = (const float*)d_in[3];  // [64]
    const float* mlp_w   = (const float*)d_in[4];  // [256,8192]
    const float* mlp_b   = (const float*)d_in[5];  // [256]
    float*       out     = (float*)d_out;          // [32,256]

    // workspace layout (bytes)
    char* ws = (char*)d_ws;
    const size_t off_a    = 0;                                   // 16 MB
    const size_t off_asum = off_a    + (size_t)32 * 64 * 4096 * 2;
    const size_t off_vlad = off_asum + (size_t)32 * 64 * 4;      // 1 MB
    const size_t off_v2   = off_vlad + (size_t)32 * 64 * 128 * 4;// 512 KB
    const size_t off_op   = off_v2   + (size_t)32 * 8192 * 2;    // 32 KB
    const size_t off_xbf  = off_op   + (size_t)32 * 256 * 4;     // 32 MB
    const size_t off_cwb  = off_xbf  + (size_t)32 * 128 * 4096 * 2; // 16 KB
    const size_t off_mwb  = off_cwb  + (size_t)64 * 128 * 2;     // 4 MB

    unsigned short* a_ws    = (unsigned short*)(ws + off_a);
    float*          asum    = (float*)(ws + off_asum);
    float*          vlad    = (float*)(ws + off_vlad);
    unsigned short* v2      = (unsigned short*)(ws + off_v2);
    float*          out_pre = (float*)(ws + off_op);
    unsigned int*   xbf     = (unsigned int*)(ws + off_xbf);
    unsigned int*   cwb     = (unsigned int*)(ws + off_cwb);
    unsigned int*   mwb     = (unsigned int*)(ws + off_mwb);

    kp_prep<<<514, 256, 0, stream>>>(conv_w, mlp_w, cwb, mwb, asum);
    k1_logits_softmax<<<dim3(128, 32), 256, 0, stream>>>(x, cwb, conv_b, a_ws, asum, xbf);
    k2_vlad<<<32, 256, 0, stream>>>(a_ws, (const unsigned short*)xbf, vlad);
    k3_finalize<<<32, 256, 0, stream>>>(vlad, asum, centers, v2);
    k4_mlp<<<32, 32, 0, stream>>>(v2, mwb, mlp_b, out_pre);
    k5_norm<<<32, 256, 0, stream>>>(out_pre, out);
}
